// GraphConv_18915035971847
// MI455X (gfx1250) — matrix operand order repaired and checked by reading, not compiled, not run on hardware
//
#include <hip/hip_runtime.h>
#include <math.h>

#define BB 8
#define CC 128
#define HH 56
#define WW 56
#define NN (HH*WW)      /* 3136 */
#define KK 8
#define KC (KK*CC)      /* 1024 */
#define NT32 (NN/32)    /* 98   */
#define ROWS 32
#define CHUNK 1568      /* columns per LDS panel chunk */
#define CTPC (CHUNK/16) /* 98 column tiles per chunk   */

typedef __attribute__((ext_vector_type(16))) __bf16 bf16x16;
typedef __attribute__((ext_vector_type(8)))  float  floatx8;

union BF16Frag { bf16x16 v; unsigned int u[8]; };

static __device__ __forceinline__ unsigned short f2bf(float f) {
    unsigned int u = __float_as_uint(f);
    u = (u + 0x7FFFu + ((u >> 16) & 1u)) >> 16;   // RNE
    return (unsigned short)u;
}
static __device__ __forceinline__ float bf2f(unsigned short h) {
    return __uint_as_float(((unsigned int)h) << 16);
}
static __device__ __forceinline__ unsigned int scale_pk(unsigned int p, float wt) {
    float lo = bf2f((unsigned short)(p & 0xffffu)) * wt;
    float hh = bf2f((unsigned short)(p >> 16)) * wt;
    return (unsigned int)f2bf(lo) | ((unsigned int)f2bf(hh) << 16);
}

// running sorted-descending top-8 insert (all static indices -> stays in VGPRs)
static __device__ __forceinline__ void topk_insert(float (&tv)[8], int (&ti)[8],
                                                   float v, int j) {
    if (v > tv[7]) {
        float cv = v; int ci = j;
#pragma unroll
        for (int s = 0; s < 8; ++s) {
            if (cv > tv[s]) {
                float t0 = tv[s]; tv[s] = cv; cv = t0;
                int   t1 = ti[s]; ti[s] = ci; ci = t1;
            }
        }
    }
}

static __device__ __forceinline__ void scan_row(const float* __restrict__ rowp,
                                                int colbase, int grow,
                                                float (&tv)[8], int (&ti)[8], int lane) {
    for (int j = lane; j < CHUNK; j += 32) {
        int gcol = colbase + j;
        if (gcol == grow) continue;          // drop self-similarity (= ref's top-1)
        topk_insert(tv, ti, rowp[j], gcol);
    }
}

// wave-parallel merge of 32 sorted 8-lists -> global top-8, then softmax+store
static __device__ __forceinline__ void emit_row(int b, int grow,
                                                float (&tv)[8], int (&ti)[8], int lane,
                                                float* __restrict__ topw,
                                                int* __restrict__ topi) {
    float outv[8]; int outi[8];
    int head = 0;
#pragma unroll
    for (int s = 0; s < 8; ++s) {
        float cv = -3.0e38f; int ci = 0x7fffffff;
#pragma unroll
        for (int q = 0; q < 8; ++q)
            if (head == q) { cv = tv[q]; ci = ti[q]; }
        int cl = lane;
#pragma unroll
        for (int d = 16; d >= 1; d >>= 1) {
            float ov = __shfl_xor(cv, d);
            int   oi = __shfl_xor(ci, d);
            int   ol = __shfl_xor(cl, d);
            bool take = (ov > cv) || (ov == cv && oi < ci);  // stable tie-break
            cv = take ? ov : cv; ci = take ? oi : ci; cl = take ? ol : cl;
        }
        if (lane == cl) head++;              // owner pops its head element
        outv[s] = cv; outi[s] = ci;
    }
    if (lane == 0) {
        float mx = outv[0], e[8], sum = 0.0f;
#pragma unroll
        for (int s = 0; s < 8; ++s) { e[s] = expf(outv[s] - mx); sum += e[s]; }
        float inv = 1.0f / sum;
        size_t base = ((size_t)b * NN + grow) * KK;
#pragma unroll
        for (int s = 0; s < 8; ++s) { topw[base + s] = e[s] * inv; topi[base + s] = outi[s]; }
    }
}

// ---------------------------------------------------------------------------
// 0) zero the accumulators we atomically add into
// ---------------------------------------------------------------------------
__global__ void init_kernel(float* __restrict__ nrm2,
                            float* __restrict__ bnsum, float* __restrict__ bnsum2) {
    int i = blockIdx.x * 256 + threadIdx.x;
    if (i < BB * NN) nrm2[i] = 0.0f;
    if (i < CC) { bnsum[i] = 0.0f; bnsum2[i] = 0.0f; }
}

// ---------------------------------------------------------------------------
// 1) (B,C,N) -> (B,N,C) bf16 transpose + per-pixel sum-of-squares partials
// ---------------------------------------------------------------------------
__global__ void prep_kernel(const float* __restrict__ x,
                            unsigned short* __restrict__ feats_bf,
                            float* __restrict__ nrm2) {
    __shared__ float tile[32][33];
    __shared__ float nsum[32];
    const int nt = blockIdx.x;
    const int ct = blockIdx.y;
    const int b  = blockIdx.z;
    const int tx = threadIdx.x;
    const int ty = threadIdx.y;
    const int n0 = nt * 32, c0 = ct * 32;
    if (ty == 0) nsum[tx] = 0.0f;
    __syncthreads();
#pragma unroll
    for (int i = 0; i < 4; ++i) {
        int c = ty + i * 8;
        tile[c][tx] = x[((size_t)b * CC + c0 + c) * NN + n0 + tx];   // coalesced in n
    }
    __syncthreads();
#pragma unroll
    for (int i = 0; i < 4; ++i) {
        int nl = ty + i * 8;
        float v = tile[tx][nl];
        feats_bf[((size_t)b * NN + n0 + nl) * CC + c0 + tx] = f2bf(v); // contig in c
        atomicAdd(&nsum[nl], v * v);
    }
    __syncthreads();
    if (ty == 0) atomicAdd(&nrm2[(size_t)b * NN + n0 + tx], nsum[tx]);
}

// ---------------------------------------------------------------------------
// 2) normalize rows -> fnorm bf16
// ---------------------------------------------------------------------------
__global__ void normalize_kernel(const unsigned short* __restrict__ feats_bf,
                                 const float* __restrict__ nrm2,
                                 unsigned short* __restrict__ fnorm_bf) {
    size_t row = blockIdx.x;
    int c = threadIdx.x;
    float inv = 1.0f / fmaxf(sqrtf(nrm2[row]), 1e-12f);
    fnorm_bf[row * CC + c] = f2bf(bf2f(feats_bf[row * CC + c]) * inv);
}

// ---------------------------------------------------------------------------
// 3) conv_w f32 -> bf16 (128 x 1024, L2-resident forever after)
// ---------------------------------------------------------------------------
__global__ void convw_kernel(const float* __restrict__ conv_w,
                             unsigned short* __restrict__ convw_bf) {
    int i = blockIdx.x * 256 + threadIdx.x;   // < 131072
    convw_bf[i] = f2bf(conv_w[i]);
}

// ---------------------------------------------------------------------------
// 4) fused similarity GEMM (bf16 WMMA) + top-8 + softmax.
//    32 rows x 3136 cols per workgroup; 32x1568 f32 LDS panel, 2 column
//    chunks; running top-8 lives in registers across chunks; wave-parallel
//    shfl-based merge. Halves L2 B-panel traffic vs 16-row tiles.
// ---------------------------------------------------------------------------
__global__ __launch_bounds__(512) void sim_topk_kernel(
        const unsigned short* __restrict__ fnorm,
        float* __restrict__ topw, int* __restrict__ topi) {
    extern __shared__ float panel[];             // [ROWS][CHUNK] = 196 KB

    const int blk   = blockIdx.x;
    const int b     = blk / NT32;
    const int rbase = (blk % NT32) * ROWS;
    const int tid   = threadIdx.x;
    const int wid   = tid >> 5;                  // 16 waves
    const int lane  = tid & 31;
    const int m     = lane & 15;
    const int hi    = lane >> 4;
    const int rsub  = wid & 1;                   // which 16-row half this wave computes
    const int cstr  = wid >> 1;                  // column-tile stripe (0..7)

    const unsigned short* fb = fnorm + (size_t)b * NN * CC;

    // Hoisted A fragments: 16 rows of this wave's half, K=0..127
    BF16Frag afrag[4];
    {
        const unsigned short* arow = fb + (size_t)(rbase + rsub * 16 + m) * CC;
#pragma unroll
        for (int kc = 0; kc < 4; ++kc) {
            int kb = kc * 32;
#pragma unroll
            for (int v = 0; v < 4; ++v) {
                afrag[kc].u[v]     = *(const unsigned int*)(arow + kb + hi * 8 + 2 * v);
                afrag[kc].u[v + 4] = *(const unsigned int*)(arow + kb + 16 + hi * 8 + 2 * v);
            }
        }
    }

    float tv0[8], tv1[8]; int ti0[8], ti1[8];
#pragma unroll
    for (int s = 0; s < 8; ++s) { tv0[s] = tv1[s] = -3.0e38f; ti0[s] = ti1[s] = 0; }

    for (int chunk = 0; chunk < 2; ++chunk) {
        const int colbase = chunk * CHUNK;
        // Phase A: fill the 32 x 1568 panel
        for (int ct = cstr; ct < CTPC; ct += 8) {
            floatx8 acc = {};
            const unsigned short* brow = fb + (size_t)(colbase + ct * 16 + m) * CC;
#pragma unroll
            for (int kc = 0; kc < 4; ++kc) {
                int kb = kc * 32;
                BF16Frag bfr;
#pragma unroll
                for (int v = 0; v < 4; ++v) {
                    bfr.u[v]     = *(const unsigned int*)(brow + kb + hi * 8 + 2 * v);
                    bfr.u[v + 4] = *(const unsigned int*)(brow + kb + 16 + hi * 8 + 2 * v);
                }
                acc = __builtin_amdgcn_wmma_f32_16x16x32_bf16(
                          false, afrag[kc].v, false, bfr.v, (short)0, acc, false, false);
            }
#pragma unroll
            for (int r = 0; r < 8; ++r)
                panel[(rsub * 16 + r + 8 * hi) * CHUNK + ct * 16 + m] = acc[r];
        }
        __syncthreads();
        // Phase B: wave 'wid' scans rows wid and wid+16 of the panel
        scan_row(panel + (size_t)wid * CHUNK,        colbase, rbase + wid,      tv0, ti0, lane);
        scan_row(panel + (size_t)(wid + 16) * CHUNK, colbase, rbase + wid + 16, tv1, ti1, lane);
        __syncthreads();
    }

    emit_row(b, rbase + wid,      tv0, ti0, lane, topw, topi);
    emit_row(b, rbase + wid + 16, tv1, ti1, lane, topw, topi);
}

// ---------------------------------------------------------------------------
// 5) gathered conv: y[b,o,n] = conv_b[o] + sum_kk W[o,kk] * w_k * feats[b,idx_k,c]
//    16-pixel tile: scaled A panel (16x1024 bf16) staged once in 32KB LDS via
//    128-bit gathers, then 8 waves x 32 bf16 WMMAs against L2-resident W.
// ---------------------------------------------------------------------------
__global__ __launch_bounds__(256) void gconv_kernel(
        const unsigned short* __restrict__ feats_bf,
        const unsigned short* __restrict__ convw_bf,
        const float* __restrict__ conv_b,
        const float* __restrict__ topw, const int* __restrict__ topi,
        float* __restrict__ ybuf) {
    __shared__ unsigned short aLds[16 * KC];     // 32 KB
    const int blk = blockIdx.x;
    const int b   = blk / (NN / 16);
    const int n0  = (blk % (NN / 16)) * 16;
    const int tid = threadIdx.x;

    {   // gather + scale: thread -> (pixel, neighbor, half-row of 64 channels)
        int mp  = tid >> 4;
        int seg = tid & 15;
        int k   = seg >> 1;
        int c0  = (seg & 1) * 64;
        size_t tb = ((size_t)b * NN + n0 + mp) * KK + k;
        int   src = topi[tb];
        float wt  = topw[tb];
        const uint4* s4 = (const uint4*)(feats_bf + ((size_t)b * NN + src) * CC + c0);
        uint4* d4 = (uint4*)(aLds + mp * KC + k * CC + c0);
#pragma unroll
        for (int t = 0; t < 8; ++t) {            // 8 x 128-bit = 64 bf16
            uint4 p = s4[t];
            p.x = scale_pk(p.x, wt); p.y = scale_pk(p.y, wt);
            p.z = scale_pk(p.z, wt); p.w = scale_pk(p.w, wt);
            d4[t] = p;
        }
    }
    __syncthreads();

    const int wid  = tid >> 5;    // 8 waves -> 8 output tiles of 16 channels
    const int lane = tid & 31;
    const int m    = lane & 15;
    const int hi   = lane >> 4;
    const int o    = wid * 16 + m;
    const unsigned short* brow = convw_bf + (size_t)o * KC;
    floatx8 acc = {};
    for (int kc = 0; kc < 32; ++kc) {
        int kb = kc * 32;
        if (kc + 1 < 32)
            __builtin_prefetch(brow + kb + 32, 0, 3);   // next Wt K-chunk from L2
        BF16Frag af, bfr;
        const unsigned short* ar = aLds + m * KC + kb;
#pragma unroll
        for (int v = 0; v < 4; ++v) {
            af.u[v]     = *(const unsigned int*)(ar + hi * 8 + 2 * v);
            af.u[v + 4] = *(const unsigned int*)(ar + 16 + hi * 8 + 2 * v);
        }
#pragma unroll
        for (int v = 0; v < 4; ++v) {
            bfr.u[v]     = *(const unsigned int*)(brow + kb + hi * 8 + 2 * v);
            bfr.u[v + 4] = *(const unsigned int*)(brow + kb + 16 + hi * 8 + 2 * v);
        }
        acc = __builtin_amdgcn_wmma_f32_16x16x32_bf16(
                  false, af.v, false, bfr.v, (short)0, acc, false, false);
    }
    float bias = conv_b[o];
#pragma unroll
    for (int r = 0; r < 8; ++r)
        ybuf[((size_t)b * CC + o) * NN + n0 + (r + 8 * hi)] = acc[r] + bias;
}

// ---------------------------------------------------------------------------
// 6) BN statistics: one block per (b,c) plane, two atomics per block
// ---------------------------------------------------------------------------
__global__ void bn_stats_kernel(const float* __restrict__ ybuf,
                                float* __restrict__ bnsum, float* __restrict__ bnsum2) {
    __shared__ float s1[256], s2[256];
    int c = blockIdx.x % CC;
    const float* p = ybuf + (size_t)blockIdx.x * NN;
    float a = 0.0f, q = 0.0f;
    for (int j = threadIdx.x; j < NN; j += 256) { float v = p[j]; a += v; q += v * v; }
    s1[threadIdx.x] = a; s2[threadIdx.x] = q;
    __syncthreads();
    for (int st = 128; st > 0; st >>= 1) {
        if (threadIdx.x < st) {
            s1[threadIdx.x] += s1[threadIdx.x + st];
            s2[threadIdx.x] += s2[threadIdx.x + st];
        }
        __syncthreads();
    }
    if (threadIdx.x == 0) { atomicAdd(&bnsum[c], s1[0]); atomicAdd(&bnsum2[c], s2[0]); }
}

// ---------------------------------------------------------------------------
// 7) BN apply + exact GELU + residual
// ---------------------------------------------------------------------------
__global__ void bn_apply_kernel(const float* __restrict__ ybuf,
                                const float* __restrict__ x,
                                const float* __restrict__ bnsum,
                                const float* __restrict__ bnsum2,
                                const float* __restrict__ gamma,
                                const float* __restrict__ beta,
                                float* __restrict__ out) {
    size_t idx = (size_t)blockIdx.x * 256 + threadIdx.x;
    const float invM = 1.0f / (float)(BB * NN);
    int c = (int)((idx / NN) % CC);
    float mean = bnsum[c] * invM;
    float var  = bnsum2[c] * invM - mean * mean;
    float rs   = rsqrtf(var + 1e-5f);
    float t    = (ybuf[idx] - mean) * rs * gamma[c] + beta[c];
    float g    = 0.5f * t * (1.0f + erff(t * 0.70710678118654752f));
    out[idx] = g + x[idx];
}

// ---------------------------------------------------------------------------
extern "C" void kernel_launch(void* const* d_in, const int* in_sizes, int n_in,
                              void* d_out, int out_size, void* d_ws, size_t ws_size,
                              hipStream_t stream) {
    const float* x      = (const float*)d_in[0];
    const float* conv_w = (const float*)d_in[1];
    const float* conv_b = (const float*)d_in[2];
    const float* gamma  = (const float*)d_in[3];
    const float* beta   = (const float*)d_in[4];
    float* out = (float*)d_out;

    char* ws = (char*)d_ws;
    size_t off = 0;
    auto take = [&](size_t bytes) { char* p = ws + off; off = (off + bytes + 255) & ~(size_t)255; return p; };
    float*          nrm2     = (float*)         take((size_t)BB * NN * 4);
    unsigned short* feats_bf = (unsigned short*)take((size_t)BB * NN * CC * 2);
    unsigned short* fnorm_bf = (unsigned short*)take((size_t)BB * NN * CC * 2);
    unsigned short* convw_bf = (unsigned short*)take((size_t)CC * KC * 2);
    float*          topw     = (float*)         take((size_t)BB * NN * KK * 4);
    int*            topi     = (int*)           take((size_t)BB * NN * KK * 4);
    float*          ybuf     = (float*)         take((size_t)BB * CC * NN * 4);
    float*          bnsum    = (float*)         take((size_t)CC * 4);
    float*          bnsum2   = (float*)         take((size_t)CC * 4);
    (void)in_sizes; (void)n_in; (void)out_size; (void)ws_size;

    init_kernel<<<(BB * NN + 255) / 256, 256, 0, stream>>>(nrm2, bnsum, bnsum2);
    prep_kernel<<<dim3(NN / 32, CC / 32, BB), dim3(32, 8), 0, stream>>>(x, feats_bf, nrm2);
    normalize_kernel<<<BB * NN, CC, 0, stream>>>(feats_bf, nrm2, fnorm_bf);
    convw_kernel<<<(CC * KC) / 256, 256, 0, stream>>>(conv_w, convw_bf);

    // 32 x 1568 f32 panel = 196 KB dynamic LDS (CDNA5: 320 KB per workgroup)
    size_t simBytes = (size_t)ROWS * CHUNK * 4;
    sim_topk_kernel<<<BB * NT32, 512, simBytes, stream>>>(fnorm_bf, topw, topi);

    gconv_kernel<<<BB * (NN / 16), 256, 0, stream>>>(feats_bf, convw_bf, conv_b, topw, topi, ybuf);
    bn_stats_kernel<<<BB * CC, 256, 0, stream>>>(ybuf, bnsum, bnsum2);
    bn_apply_kernel<<<(BB * CC * NN) / 256, 256, 0, stream>>>(ybuf, x, bnsum, bnsum2, gamma, beta, out);
}
